// Zero_Parameter_32830730011392
// MI455X (gfx1250) — matrix operand order repaired and checked by reading, not compiled, not run on hardware
//
#include <hip/hip_runtime.h>
#include <hip/hip_bf16.h>

// ---------------------------------------------------------------------------
// MI455X (gfx1250) cross-attention pipeline, bf16 WMMA with f32 accumulation.
// FLOPs ~71G, bytes ~120MB -> matrix-core bound -> v_wmma_f32_16x16x32_bf16.
// Round 2: Tensor Data Mover (tensor_load_to_lds) staging for bf16 tiles
// (attention Q/K tiles + out-projection A tile), TENSORcnt-synchronized.
// ---------------------------------------------------------------------------

typedef __attribute__((ext_vector_type(16))) __bf16 v16bf;
typedef __attribute__((ext_vector_type(8)))  float  v8f;

#define DIM_T   768
#define DIM_V   1024
#define DIM_OUT 768
#define NHEADS  8
#define HEADD   96
#define BATCH   8
#define NTOK    1024
#define MROWS   (BATCH * NTOK)   // 8192

#if __has_builtin(__builtin_amdgcn_tensor_load_to_lds)
#define HAVE_TDM 1
#else
#define HAVE_TDM 0
#endif

// ---- helpers ---------------------------------------------------------------

// f32 -> bf16 round-to-nearest-even via bit ops (no reliance on __bf16 cvt).
static __device__ inline __bf16 f2bf(float f) {
  union { float f; unsigned int u; } x; x.f = f;
  unsigned int r = x.u + 0x7FFFu + ((x.u >> 16) & 1u);
  unsigned short h = (unsigned short)(r >> 16);
  __bf16 out;
  __builtin_memcpy(&out, &h, 2);
  return out;
}

// Build a 16-element bf16 fragment from two contiguous 8-element LDS runs.
static __device__ inline v16bf ld16(const __bf16* p0, const __bf16* p1) {
  v16bf r;
#pragma unroll
  for (int i = 0; i < 8; ++i) { r[i] = p0[i]; r[i + 8] = p1[i]; }
  return r;
}

static __device__ inline v8f wmma_bf16(v16bf a, v16bf b, v8f c) {
  // (neg_a, A, neg_b, B, c_mod, C, reuse_a, reuse_b)
  return __builtin_amdgcn_wmma_f32_16x16x32_bf16(false, a, false, b,
                                                 (short)0, c, false, false);
}

#if HAVE_TDM
typedef __attribute__((ext_vector_type(4))) unsigned int u32x4;
typedef __attribute__((ext_vector_type(8))) int i32x8;
typedef __attribute__((ext_vector_type(4))) int i32x4;

// Generic shared pointer: addr[31:0] is the LDS byte offset (LDS aperture).
static __device__ inline unsigned lds_off(const void* p) {
  return (unsigned)(unsigned long long)(uintptr_t)p;
}

// TDM 2D bf16 tile load: rows x cols tile, row stride in elements, packed
// contiguously into LDS (row length * 2B per row). D# per CDNA5 ISA 8.3/8.4.
static __device__ inline void tdm_load_2d_bf16(const __bf16* gsrc,
                                               unsigned lds_byte,
                                               unsigned rows, unsigned cols,
                                               unsigned stride_elems) {
  unsigned long long ga = (unsigned long long)(uintptr_t)gsrc;
  u32x4 g0;
  g0[0] = 1u;                                   // count=1, user descriptor
  g0[1] = lds_byte;                             // lds_addr
  g0[2] = (unsigned)(ga & 0xFFFFFFFFu);         // global_addr[31:0]
  g0[3] = (unsigned)((ga >> 32) & 0x1FFFFFFu)   // global_addr[56:32]
          | (2u << 30);                         // type=2 ("image")
  i32x8 g1;
  g1[0] = (int)(1u << 16);                      // data_size=2B, no multicast
  g1[1] = (int)((cols & 0xFFFFu) << 16);        // tensor_dim0[15:0]
  g1[2] = (int)(((cols >> 16) & 0xFFFFu) |      // tensor_dim0[31:16]
                ((rows & 0xFFFFu) << 16));      // tensor_dim1[15:0]
  g1[3] = (int)(((rows >> 16) & 0xFFFFu) |      // tensor_dim1[31:16]
                ((cols & 0xFFFFu) << 16));      // tile_dim0
  g1[4] = (int)(rows & 0xFFFFu);                // tile_dim1 (tile_dim2=0)
  g1[5] = (int)stride_elems;                    // tensor_dim0_stride[31:0]
  g1[6] = 0;                                    // stride hi / dim1_stride lo
  g1[7] = 0;
  i32x4 z = {0, 0, 0, 0};
#if defined(__clang_major__) && (__clang_major__ >= 23)
  i32x8 z8 = {0, 0, 0, 0, 0, 0, 0, 0};
  __builtin_amdgcn_tensor_load_to_lds(g0, g1, z, z, z8, 0);
#else
  __builtin_amdgcn_tensor_load_to_lds(g0, g1, z, z, 0);
#endif
}

static __device__ inline void tdm_wait0() {
#if __has_builtin(__builtin_amdgcn_s_wait_tensorcnt)
  __builtin_amdgcn_s_wait_tensorcnt(0);
#else
  asm volatile("s_wait_tensorcnt 0x0" ::: "memory");
#endif
}
#endif  // HAVE_TDM

// ---------------------------------------------------------------------------
// Tiled GEMM with bias: C[M,N] = A[M,K] * W[K,N] + bias.
// A is f32 (A_BF16=false) or bf16 (true); C is bf16 (OUT_F32=false) or f32.
// WG = 256 threads = 8 waves, tile 64(M) x 128(N), BK = 32.
// Wave grid 2(M) x 4(N); each wave computes 32x32 via 4 WMMA per K-step.
// bf16 A tiles are staged by the Tensor Data Mover when available.
// ---------------------------------------------------------------------------
template <bool A_BF16, bool OUT_F32>
__global__ __launch_bounds__(256) void gemm_bias_kernel(
    const void* __restrict__ Aptr, const float* __restrict__ W,
    const float* __restrict__ bias, void* __restrict__ Cptr,
    int M, int N, int K) {
  __shared__ __bf16 As[64 * 32];   // [m][k], stride 32
  __shared__ __bf16 Bs[128 * 32];  // [n][k] (k-contiguous for B fragments)

  const int t    = threadIdx.x;
  const int lane = t & 31;
  const int wv   = t >> 5;
  const int wm   = wv & 1;   // 0..1 -> 32-row slab
  const int wn   = wv >> 1;  // 0..3 -> 32-col slab
  const int m0   = blockIdx.y * 64;
  const int n0   = blockIdx.x * 128;
  const int half = lane >> 4;
  const int ln   = lane & 15;

  v8f acc[2][2];
#pragma unroll
  for (int i = 0; i < 2; ++i)
#pragma unroll
    for (int j = 0; j < 2; ++j)
#pragma unroll
      for (int r = 0; r < 8; ++r) acc[i][j][r] = 0.0f;

  const int arow = t >> 2;        // 0..63
  const int acol = (t & 3) * 8;   // 0,8,16,24
  const int brow = t >> 3;        // 0..31 (k)
  const int bcol = (t & 7) * 16;  // 0..112 (n)

  for (int k0 = 0; k0 < K; k0 += 32) {
    __syncthreads();
    // ---- stage A tile (64x32) as bf16 --------------------------------
    if (A_BF16) {
#if HAVE_TDM
      if (wv == 0)
        tdm_load_2d_bf16((const __bf16*)Aptr + (size_t)m0 * K + k0,
                         lds_off(As), 64, 32, (unsigned)K);
#else
      const __bf16* A = (const __bf16*)Aptr;
      const __bf16* src = A + (size_t)(m0 + arow) * K + k0 + acol;
      *(uint4*)(&As[arow * 32 + acol]) = *(const uint4*)src;
#endif
    } else {
      const float* A = (const float*)Aptr;
      const float* src = A + (size_t)(m0 + arow) * K + k0 + acol;
      float4 f0 = *(const float4*)(src);
      float4 f1 = *(const float4*)(src + 4);
      __bf16* dst = &As[arow * 32 + acol];
      dst[0] = f2bf(f0.x); dst[1] = f2bf(f0.y);
      dst[2] = f2bf(f0.z); dst[3] = f2bf(f0.w);
      dst[4] = f2bf(f1.x); dst[5] = f2bf(f1.y);
      dst[6] = f2bf(f1.z); dst[7] = f2bf(f1.w);
    }
    // ---- stage B tile (32x128) transposed to [n][k] ------------------
    {
      const float* src = W + (size_t)(k0 + brow) * N + n0 + bcol;
      if (k0 + 32 < K)
        __builtin_prefetch((const char*)(src + (size_t)32 * N), 0, 1);
#pragma unroll
      for (int i = 0; i < 16; i += 4) {
        float4 f = *(const float4*)(src + i);
        Bs[(bcol + i + 0) * 32 + brow] = f2bf(f.x);
        Bs[(bcol + i + 1) * 32 + brow] = f2bf(f.y);
        Bs[(bcol + i + 2) * 32 + brow] = f2bf(f.z);
        Bs[(bcol + i + 3) * 32 + brow] = f2bf(f.w);
      }
    }
#if HAVE_TDM
    if (A_BF16 && wv == 0) tdm_wait0();
#endif
    __syncthreads();

    // ---- fragments + WMMA --------------------------------------------
    v16bf afr[2], bfr[2];
#pragma unroll
    for (int i = 0; i < 2; ++i) {
      // A layout: lane<16 -> k {0..7,16..23}; lane>=16 -> {8..15,24..31}
      const __bf16* p = &As[(wm * 32 + i * 16 + ln) * 32 + half * 8];
      afr[i] = ld16(p, p + 16);
    }
#pragma unroll
    for (int j = 0; j < 2; ++j) {
      // B layout: lane<16 -> k {0..7,16..23}; lane>=16 -> {8..15,24..31}
      const __bf16* p = &Bs[(wn * 32 + j * 16 + ln) * 32 + half * 8];
      bfr[j] = ld16(p, p + 16);
    }
#pragma unroll
    for (int i = 0; i < 2; ++i)
#pragma unroll
      for (int j = 0; j < 2; ++j)
        acc[i][j] = wmma_bf16(afr[i], bfr[j], acc[i][j]);
  }

  // ---- epilogue: bias add, store -------------------------------------
#pragma unroll
  for (int i = 0; i < 2; ++i) {
#pragma unroll
    for (int j = 0; j < 2; ++j) {
      const int col = n0 + wn * 32 + j * 16 + ln;
      const float bv = bias[col];
#pragma unroll
      for (int r = 0; r < 8; ++r) {
        const int row = m0 + wm * 32 + i * 16 + r + 8 * half;
        const float v = acc[i][j][r] + bv;
        if (OUT_F32)
          ((float*)Cptr)[(size_t)row * N + col] = v;
        else
          ((__bf16*)Cptr)[(size_t)row * N + col] = f2bf(v);
      }
    }
  }
}

// ---------------------------------------------------------------------------
// Flash attention: WG = 128 threads (4 waves); one WG per (b, h, 64 q-rows).
// Each wave owns 16 q-rows; streams 32-key K/V blocks; online softmax with
// 16-lane-half shuffle reductions matching the WMMA C-matrix layout.
// Q tile and K blocks staged by the Tensor Data Mover (wave 0 issues the
// descriptor, s_wait_tensorcnt 0, then the workgroup barrier publishes LDS).
// V needs a transpose for the PV B-fragment, so it stays on the manual path.
// ---------------------------------------------------------------------------
__global__ __launch_bounds__(128) void attn_kernel(
    const __bf16* __restrict__ Q, const __bf16* __restrict__ Kb,
    const __bf16* __restrict__ Vb, __bf16* __restrict__ ctx) {
  __shared__ __bf16 Qs[64 * 96];     // [qrow][c]
  __shared__ __bf16 Ks[32 * 96];     // [key][c]
  __shared__ __bf16 Vs[96 * 32];     // [c][key] (transposed for PV B-frag)
  __shared__ __bf16 Ps[4][16 * 32];  // per-wave P staging (C->A relayout)

  const int t     = threadIdx.x;
  const int lane  = t & 31;
  const int wv    = t >> 5;
  const int half  = lane >> 4;
  const int ln    = lane & 15;
  const int qbase = blockIdx.x * 64;
  const int h     = blockIdx.y;
  const int b     = blockIdx.z;

  // ---- load Q tile 64x96 (bf16) once --------------------------------
#if HAVE_TDM
  if (wv == 0) {
    const __bf16* src =
        Q + ((size_t)(b * NTOK + qbase)) * DIM_OUT + h * HEADD;
    tdm_load_2d_bf16(src, lds_off(Qs), 64, 96, DIM_OUT);
  }
#else
  {
    const int row = t >> 1;
    const int c0  = (t & 1) * 48;
    const __bf16* src =
        Q + ((size_t)(b * NTOK + qbase + row)) * DIM_OUT + h * HEADD + c0;
    __bf16* dst = &Qs[row * 96 + c0];
#pragma unroll
    for (int i = 0; i < 48; i += 8)
      *(uint4*)(dst + i) = *(const uint4*)(src + i);
  }
#endif

  float m[8], l[8];
#pragma unroll
  for (int r = 0; r < 8; ++r) { m[r] = -1e30f; l[r] = 0.0f; }
  v8f oacc[6];
#pragma unroll
  for (int cg = 0; cg < 6; ++cg)
#pragma unroll
    for (int r = 0; r < 8; ++r) oacc[cg][r] = 0.0f;

  const int kvrow = t >> 2;        // 0..31
  const int kvc0  = (t & 3) * 24;  // 0,24,48,72

  for (int kb = 0; kb < NTOK; kb += 32) {
    __syncthreads();
    // ---- stage K (TDM, row-major) and V (manual, transposed) ---------
#if HAVE_TDM
    if (wv == 0)
      tdm_load_2d_bf16(Kb + ((size_t)(b * NTOK + kb)) * DIM_OUT + h * HEADD,
                       lds_off(Ks), 32, 96, DIM_OUT);
#else
    {
      const __bf16* srck =
          Kb + ((size_t)(b * NTOK + kb + kvrow)) * DIM_OUT + h * HEADD + kvc0;
      __bf16* dstk = &Ks[kvrow * 96 + kvc0];
#pragma unroll
      for (int i = 0; i < 24; i += 8)
        *(uint4*)(dstk + i) = *(const uint4*)(srck + i);
    }
#endif
    {
      const __bf16* srcv =
          Vb + ((size_t)(b * NTOK + kb + kvrow)) * DIM_OUT + h * HEADD + kvc0;
#pragma unroll
      for (int i = 0; i < 24; ++i)
        Vs[(kvc0 + i) * 32 + kvrow] = srcv[i];
    }
#if HAVE_TDM
    if (wv == 0) tdm_wait0();  // covers Q TDM on first iteration (in-order)
#endif
    __syncthreads();

    // ---- S = Q K^T : two 16x16 tiles, 3 k-chunks of 32 over HEADD=96 -
    v8f s0, s1;
#pragma unroll
    for (int r = 0; r < 8; ++r) { s0[r] = 0.0f; s1[r] = 0.0f; }
#pragma unroll
    for (int cc = 0; cc < 3; ++cc) {
      const __bf16* ap = &Qs[(wv * 16 + ln) * 96 + cc * 32 + half * 8];
      v16bf a = ld16(ap, ap + 16);
      const __bf16* bp0 = &Ks[(0 * 16 + ln) * 96 + cc * 32 + half * 8];
      const __bf16* bp1 = &Ks[(1 * 16 + ln) * 96 + cc * 32 + half * 8];
      s0 = wmma_bf16(a, ld16(bp0, bp0 + 16), s0);
      s1 = wmma_bf16(a, ld16(bp1, bp1 + 16), s1);
    }

    // ---- online softmax (row reductions across each 16-lane half) ----
    float sc[8];
#pragma unroll
    for (int r = 0; r < 8; ++r) {
      float mx = fmaxf(s0[r], s1[r]);
      mx = fmaxf(mx, __shfl_xor(mx, 1, 32));
      mx = fmaxf(mx, __shfl_xor(mx, 2, 32));
      mx = fmaxf(mx, __shfl_xor(mx, 4, 32));
      mx = fmaxf(mx, __shfl_xor(mx, 8, 32));
      const float nm = fmaxf(m[r], mx);
      sc[r] = __expf(m[r] - nm);
      m[r] = nm;
      s0[r] = __expf(s0[r] - nm);
      s1[r] = __expf(s1[r] - nm);
      float sm = s0[r] + s1[r];
      sm += __shfl_xor(sm, 1, 32);
      sm += __shfl_xor(sm, 2, 32);
      sm += __shfl_xor(sm, 4, 32);
      sm += __shfl_xor(sm, 8, 32);
      l[r] = l[r] * sc[r] + sm;
    }
#pragma unroll
    for (int cg = 0; cg < 6; ++cg)
#pragma unroll
      for (int r = 0; r < 8; ++r) oacc[cg][r] *= sc[r];

    // ---- restage P (C layout -> A layout) through per-wave LDS -------
    // Intra-wave DS ops are in-order on CDNA5: no barrier needed.
    __bf16* P = Ps[wv];
#pragma unroll
    for (int r = 0; r < 8; ++r) {
      P[(r + 8 * half) * 32 + ln]      = f2bf(s0[r]);
      P[(r + 8 * half) * 32 + 16 + ln] = f2bf(s1[r]);
    }
    const __bf16* pa = &P[ln * 32 + half * 8];
    v16bf aP = ld16(pa, pa + 16);
#pragma unroll
    for (int cg = 0; cg < 6; ++cg) {
      const __bf16* vp = &Vs[(cg * 16 + ln) * 32 + half * 8];
      oacc[cg] = wmma_bf16(aP, ld16(vp, vp + 16), oacc[cg]);
    }
  }

  // ---- normalize by l and store context (bf16) -----------------------
#pragma unroll
  for (int cg = 0; cg < 6; ++cg) {
#pragma unroll
    for (int r = 0; r < 8; ++r) {
      const int row = qbase + wv * 16 + r + 8 * half;
      const float v = oacc[cg][r] / l[r];
      ctx[((size_t)(b * NTOK + row)) * DIM_OUT + h * HEADD + cg * 16 + ln] =
          f2bf(v);
    }
  }
}

// ---------------------------------------------------------------------------
// Per-row L2 normalize; writes both tuple outputs.
// ---------------------------------------------------------------------------
__global__ __launch_bounds__(256) void l2norm_kernel(
    const float* __restrict__ X, float* __restrict__ out, int M, int N) {
  const int row = blockIdx.x;
  const float* x = X + (size_t)row * N;
  float ss = 0.0f;
  for (int c = threadIdx.x; c < N; c += blockDim.x) {
    const float v = x[c];
    ss += v * v;
  }
#pragma unroll
  for (int off = 16; off > 0; off >>= 1) ss += __shfl_xor(ss, off, 32);
  __shared__ float red[8];
  const int lane = threadIdx.x & 31, w = threadIdx.x >> 5;
  if (lane == 0) red[w] = ss;
  __syncthreads();
  if (w == 0) {
    float tt = (lane < 8) ? red[lane] : 0.0f;
#pragma unroll
    for (int off = 4; off > 0; off >>= 1) tt += __shfl_xor(tt, off, 32);
    if (lane == 0) red[0] = tt;
  }
  __syncthreads();
  const float inv = rsqrtf(red[0]);
  float* o1 = out + (size_t)row * N;
  float* o2 = out + (size_t)M * N + (size_t)row * N;
  for (int c = threadIdx.x; c < N; c += blockDim.x) {
    const float v = x[c] * inv;
    o1[c] = v;
    o2[c] = v;
  }
}

// ---------------------------------------------------------------------------
extern "C" void kernel_launch(void* const* d_in, const int* in_sizes, int n_in,
                              void* d_out, int out_size, void* d_ws,
                              size_t ws_size, hipStream_t stream) {
  (void)in_sizes; (void)n_in; (void)out_size; (void)ws_size;
  const float* F_t = (const float*)d_in[0];
  const float* F_s = (const float*)d_in[1];
  const float* Wq  = (const float*)d_in[2];
  const float* bq  = (const float*)d_in[3];
  const float* Wk  = (const float*)d_in[4];
  const float* bk  = (const float*)d_in[5];
  const float* Wv  = (const float*)d_in[6];
  const float* bv  = (const float*)d_in[7];
  const float* Wp  = (const float*)d_in[8];
  const float* bp  = (const float*)d_in[9];

  const size_t MAT = (size_t)MROWS * DIM_OUT;  // 8192*768
  __bf16* qb  = (__bf16*)d_ws;
  __bf16* kbf = qb + MAT;
  __bf16* vbf = kbf + MAT;
  __bf16* ctx = vbf + MAT;
  float*  proj = (float*)(ctx + MAT);

  const dim3 ggrid(DIM_OUT / 128, MROWS / 64);  // (6, 128)

  // Q/K/V projections: f32 in -> bf16 out, fused f32->bf16 staging.
  gemm_bias_kernel<false, false><<<ggrid, 256, 0, stream>>>(
      (const void*)F_t, Wq, bq, (void*)qb, MROWS, DIM_OUT, DIM_T);
  gemm_bias_kernel<false, false><<<ggrid, 256, 0, stream>>>(
      (const void*)F_s, Wk, bk, (void*)kbf, MROWS, DIM_OUT, DIM_V);
  gemm_bias_kernel<false, false><<<ggrid, 256, 0, stream>>>(
      (const void*)F_s, Wv, bv, (void*)vbf, MROWS, DIM_OUT, DIM_V);

  // Flash attention per (b, h, 64 q-rows).
  attn_kernel<<<dim3(NTOK / 64, NHEADS, BATCH), 128, 0, stream>>>(qb, kbf, vbf,
                                                                  ctx);

  // Output projection: bf16 in (TDM-staged) -> f32 out.
  gemm_bias_kernel<true, true><<<ggrid, 256, 0, stream>>>(
      (const void*)ctx, Wp, bp, (void*)proj, MROWS, DIM_OUT, DIM_OUT);

  // L2 normalize; write tuple (out, out).
  l2norm_kernel<<<MROWS, 256, 0, stream>>>(proj, (float*)d_out, MROWS, DIM_OUT);
}
